// TransformerBlock_31233002176577
// MI455X (gfx1250) — matrix operand order repaired and checked by reading, not compiled, not run on hardware
//
#include <hip/hip_runtime.h>

// ---------------------------------------------------------------------------
// Types / helpers
// ---------------------------------------------------------------------------
typedef unsigned short u16;
typedef __attribute__((ext_vector_type(16))) __bf16 v16bf;
typedef __attribute__((ext_vector_type(8)))  __bf16 v8bf;
typedef __attribute__((ext_vector_type(8)))  float  v8f;
typedef __attribute__((ext_vector_type(4)))  int    v4i;

#define AS1 __attribute__((address_space(1)))
#define AS3 __attribute__((address_space(3)))
#define LOG2E 1.44269504088896340736f

static __device__ __forceinline__ u16 f2b(float f) {
  unsigned int u = __float_as_uint(f);
  u += 0x7fffu + ((u >> 16) & 1u);          // round-to-nearest-even bf16
  return (u16)(u >> 16);
}

// A/B fragment loaders. CDNA5 16-bit WMMA operand layouts (ISA 7.12.2):
//  A (16x32, MxK): lane<16 row M=lane holds K = 8*(lane>=16)+0..7 in elems 0..7
//                  and K = 16+8*(lane>=16)+0..7 in elems 8..15  -> two 16B loads
//  B (32x16, KxN): lane%16 = column N, elems 0..15 hold K in the same order as A
//                  -> two 16B loads; contiguous 32B loads only where A is contiguous too
static __device__ __forceinline__ v16bf ld_frag_pair(const u16* p0, const u16* p1) {
  v8bf lo = *reinterpret_cast<const v8bf*>(p0);
  v8bf hi = *reinterpret_cast<const v8bf*>(p1);
  v16bf r;
#pragma unroll
  for (int i = 0; i < 8; ++i) { r[i] = lo[i]; r[i + 8] = hi[i]; }
  return r;
}
static __device__ __forceinline__ v16bf ld_frag32B(const u16* p) {
  return *reinterpret_cast<const v16bf*>(p);
}
static __device__ __forceinline__ v8f wmma_bf16(v16bf a, v16bf b, v8f c) {
  return __builtin_amdgcn_wmma_f32_16x16x32_bf16(false, a, false, b, (short)0, c,
                                                 false, false);
}

// ---------------------------------------------------------------------------
// fp32 -> bf16 conversion (vectorized, n multiple of 4)
// ---------------------------------------------------------------------------
__global__ __launch_bounds__(256) void f2bf_kernel(const float* __restrict__ in,
                                                   u16* __restrict__ out, int n4) {
  int i = blockIdx.x * 256 + threadIdx.x;
  if (i >= n4) return;
  float4 v = reinterpret_cast<const float4*>(in)[i];
  union { u16 r[4]; uint2 u; } pk;
  pk.r[0] = f2b(v.x); pk.r[1] = f2b(v.y); pk.r[2] = f2b(v.z); pk.r[3] = f2b(v.w);
  reinterpret_cast<uint2*>(out)[i] = pk.u;
}

// ---------------------------------------------------------------------------
// GEMM: C[M,N] = A[M,K] @ W[N,K]^T   (bf16 in, f32 accumulate via WMMA)
// Block tile 128x128, 8 waves (4m x 2n), wave tile 32x64, K-step 32.
// Double-buffered LDS tiles filled with GLOBAL_LOAD_ASYNC_TO_LDS_B128
// (ASYNCcnt path): one s_wait_asynccnt + one barrier per K-step, no VGPR
// staging. LDS pitch = 40 halves (80B) to spread banks; 16B-aligned chunks.
// MODE 0: QKV scatter (q scaled 0.125; v stored dim-major)  MODE 1: +bias -> f32
// MODE 2: +bias, PReLU -> bf16
// ---------------------------------------------------------------------------
template <int MODE>
__global__ __launch_bounds__(256) void gemm_bf16_kernel(
    const u16* __restrict__ A, const u16* __restrict__ W, int K, int N,
    const float* __restrict__ bias, const float* __restrict__ prelu,
    float* __restrict__ outF, u16* __restrict__ outB,
    u16* __restrict__ qb, u16* __restrict__ kb, u16* __restrict__ vtb) {
  __shared__ u16 lA[2][128 * 40];
  __shared__ u16 lB[2][128 * 40];
  const int tid = threadIdx.x;
  const int lane = tid & 31;
  const int wid = tid >> 5;
  const int wm = wid & 3;          // 32-row slice within block
  const int wn = wid >> 2;         // 64-col slice within block
  const int hi = lane >> 4;
  const int l15 = lane & 15;
  const int kh8 = hi * 8;
  const int kh16 = hi * 16;
  const int m0g = blockIdx.y * 128;
  const int n0g = blockIdx.x * 128;

  // this thread's two 16B chunks of each tile
  const int row0 = tid >> 2, ch0 = tid & 3;
  const int row1 = (tid + 256) >> 2, ch1 = (tid + 256) & 3;

  auto issue_tile = [&](int k0, int buf) {
    __builtin_amdgcn_global_load_async_to_lds_b128(
        (AS1 v4i*)(A + (size_t)(m0g + row0) * K + k0 + ch0 * 8),
        (AS3 v4i*)&lA[buf][row0 * 40 + ch0 * 8], 0, 0);
    __builtin_amdgcn_global_load_async_to_lds_b128(
        (AS1 v4i*)(A + (size_t)(m0g + row1) * K + k0 + ch1 * 8),
        (AS3 v4i*)&lA[buf][row1 * 40 + ch1 * 8], 0, 0);
    __builtin_amdgcn_global_load_async_to_lds_b128(
        (AS1 v4i*)(W + (size_t)(n0g + row0) * K + k0 + ch0 * 8),
        (AS3 v4i*)&lB[buf][row0 * 40 + ch0 * 8], 0, 0);
    __builtin_amdgcn_global_load_async_to_lds_b128(
        (AS1 v4i*)(W + (size_t)(n0g + row1) * K + k0 + ch1 * 8),
        (AS3 v4i*)&lB[buf][row1 * 40 + ch1 * 8], 0, 0);
  };

  const v8f z = {0.f, 0.f, 0.f, 0.f, 0.f, 0.f, 0.f, 0.f};
  v8f acc[2][4];
#pragma unroll
  for (int i = 0; i < 2; ++i)
#pragma unroll
    for (int j = 0; j < 4; ++j) acc[i][j] = z;

  issue_tile(0, 0);                      // prologue: fill buffer 0
  int buf = 0;
  for (int k0 = 0; k0 < K; k0 += 32) {
    asm volatile("s_wait_asynccnt 0x0" ::: "memory");  // own tile landed in LDS
    __syncthreads();                                   // everyone's tile landed
    if (k0 + 32 < K) issue_tile(k0 + 32, buf ^ 1);     // prefetch next tile

    v16bf af[2];
#pragma unroll
    for (int fm = 0; fm < 2; ++fm) {
      const u16* p = &lA[buf][(wm * 32 + fm * 16 + l15) * 40 + kh8];
      af[fm] = ld_frag_pair(p, p + 16);
    }
#pragma unroll
    for (int fn = 0; fn < 4; ++fn) {
      const u16* p = &lB[buf][(wn * 64 + fn * 16 + l15) * 40 + kh8];
      v16bf bf = ld_frag_pair(p, p + 16);
#pragma unroll
      for (int fm = 0; fm < 2; ++fm) acc[fm][fn] = wmma_bf16(af[fm], bf, acc[fm][fn]);
    }
    buf ^= 1;
  }

  // Epilogue. C frag layout: elem e -> row = e + 8*(lane>=16), col = lane&15.
#pragma unroll
  for (int fm = 0; fm < 2; ++fm) {
#pragma unroll
    for (int fn = 0; fn < 4; ++fn) {
#pragma unroll
      for (int e = 0; e < 8; ++e) {
        int gm = m0g + wm * 32 + fm * 16 + e + 8 * hi;
        int gn = n0g + wn * 64 + fn * 16 + l15;
        float v = acc[fm][fn][e];
        if (MODE == 0) {
          int which = gn >> 10, c1 = gn & 1023;
          int h = c1 >> 6, d = c1 & 63;
          int b = gm >> 11, n = gm & 2047;
          size_t bh = (size_t)(b * 16 + h);
          if (which == 0)      qb[(bh * 2048 + n) * 64 + d] = f2b(v * 0.125f);
          else if (which == 1) kb[(bh * 2048 + n) * 64 + d] = f2b(v);
          else                 vtb[(bh * 64 + d) * 2048 + n] = f2b(v);   // dim-major V
        } else if (MODE == 1) {
          outF[(size_t)gm * N + gn] = v + bias[gn];
        } else {
          float t = v + bias[gn];
          float a = prelu[0];
          t = t >= 0.f ? t : a * t;
          outB[(size_t)gm * N + gn] = f2b(t);
        }
      }
    }
  }
}

// ---------------------------------------------------------------------------
// Flash attention: grid (B*H, N/128), 8 waves x 16 q-rows, 32-key tiles.
// S = q@k^T via 4 WMMAs, online softmax (shfl_xor row reduce, exp2),
// P repacked C->A layout through per-wave LDS, O += P@V via 4 WMMAs.
// ---------------------------------------------------------------------------
__global__ __launch_bounds__(256) void attn_kernel(const u16* __restrict__ q,
                                                   const u16* __restrict__ k,
                                                   const u16* __restrict__ vt,
                                                   u16* __restrict__ o) {
  __shared__ u16 psh[8][16 * 40];            // per-wave P scratch (pitch 40)
  const int tid = threadIdx.x, lane = tid & 31, wid = tid >> 5;
  const int hi = lane >> 4, l15 = lane & 15;
  const int kh8 = hi * 8, kh16 = hi * 16;
  const int bh = blockIdx.x;
  const int bq = bh >> 4, hh = bh & 15;
  const int q0 = blockIdx.y * 128 + wid * 16;
  const u16* qp = q + (size_t)bh * 2048 * 64;
  const u16* kp = k + (size_t)bh * 2048 * 64;
  const u16* vp = vt + (size_t)bh * 64 * 2048;

  v16bf qa[2];
#pragma unroll
  for (int kk = 0; kk < 2; ++kk) {
    const u16* p = qp + (size_t)(q0 + l15) * 64 + kk * 32 + kh16;
    qa[kk] = ld_frag32B(p);
  }

  const v8f z = {0.f, 0.f, 0.f, 0.f, 0.f, 0.f, 0.f, 0.f};
  v8f oacc[4];
#pragma unroll
  for (int fd = 0; fd < 4; ++fd) oacc[fd] = z;
  float rmax[8], rsum[8];
#pragma unroll
  for (int e = 0; e < 8; ++e) { rmax[e] = -3.0e38f; rsum[e] = 0.f; }

  u16* myp = &psh[wid][0];

  for (int kt = 0; kt < 64; ++kt) {
    const int key0 = kt * 32;
    v8f s0 = z, s1 = z;
    {
      const u16* pk0 = kp + (size_t)(key0 + l15) * 64 + kh16;        // keys 0..15
      const u16* pk1 = kp + (size_t)(key0 + 16 + l15) * 64 + kh16;   // keys 16..31
      s0 = wmma_bf16(qa[0], ld_frag32B(pk0), s0);
      s0 = wmma_bf16(qa[1], ld_frag32B(pk0 + 32), s0);
      s1 = wmma_bf16(qa[0], ld_frag32B(pk1), s1);
      s1 = wmma_bf16(qa[1], ld_frag32B(pk1 + 32), s1);
    }
    float cs[8];
#pragma unroll
    for (int e = 0; e < 8; ++e) {
      float m2 = fmaxf(s0[e], s1[e]);
#pragma unroll
      for (int d = 8; d >= 1; d >>= 1) m2 = fmaxf(m2, __shfl_xor(m2, d, 32));
      float nm = fmaxf(rmax[e], m2);
      cs[e] = exp2f((rmax[e] - nm) * LOG2E);
      rmax[e] = nm;
      float p0 = exp2f((s0[e] - nm) * LOG2E);
      float p1 = exp2f((s1[e] - nm) * LOG2E);
      float ps = p0 + p1;
#pragma unroll
      for (int d = 8; d >= 1; d >>= 1) ps += __shfl_xor(ps, d, 32);
      rsum[e] = rsum[e] * cs[e] + ps;
      int r = e + 8 * hi;
      myp[r * 40 + l15] = f2b(p0);
      myp[r * 40 + 16 + l15] = f2b(p1);
    }
#pragma unroll
    for (int fd = 0; fd < 4; ++fd)
#pragma unroll
      for (int e = 0; e < 8; ++e) oacc[fd][e] *= cs[e];

    asm volatile("s_wait_dscnt 0x0" ::: "memory");   // P stores visible to re-load
    v16bf pa;
    {
      const u16* pp = myp + l15 * 40 + kh16;
      pa = ld_frag_pair(pp, pp + 8);
    }
#pragma unroll
    for (int fd = 0; fd < 4; ++fd) {
      const u16* pv = vp + (size_t)(fd * 16 + l15) * 2048 + key0 + kh16;
      oacc[fd] = wmma_bf16(pa, ld_frag32B(pv), oacc[fd]);
    }
  }
#pragma unroll
  for (int fd = 0; fd < 4; ++fd) {
#pragma unroll
    for (int e = 0; e < 8; ++e) {
      int row = q0 + e + 8 * hi;
      int token = bq * 2048 + row;
      float val = oacc[fd][e] / rsum[e];
      o[(size_t)token * 1024 + hh * 64 + fd * 16 + l15] = f2b(val);
    }
  }
}

// ---------------------------------------------------------------------------
// Fused residual + LayerNorm over rows of 1024 (one block per row).
// outF (f32) and/or outB (bf16) written when non-null.
// ---------------------------------------------------------------------------
__global__ __launch_bounds__(256) void ln_kernel(const float* __restrict__ ra,
                                                 const float* __restrict__ rb,
                                                 const float* __restrict__ g,
                                                 const float* __restrict__ be,
                                                 float* __restrict__ outF,
                                                 u16* __restrict__ outB) {
  __shared__ float red[2][256];
  const int row = blockIdx.x, t = threadIdx.x;
  const size_t base = (size_t)row * 1024 + t * 4;
  float4 a = *reinterpret_cast<const float4*>(ra + base);
  float4 b = *reinterpret_cast<const float4*>(rb + base);
  float s0 = a.x + b.x, s1 = a.y + b.y, s2 = a.z + b.z, s3 = a.w + b.w;
  red[0][t] = s0 + s1 + s2 + s3;
  red[1][t] = s0 * s0 + s1 * s1 + s2 * s2 + s3 * s3;
  __syncthreads();
  for (int off = 128; off > 0; off >>= 1) {
    if (t < off) { red[0][t] += red[0][t + off]; red[1][t] += red[1][t + off]; }
    __syncthreads();
  }
  float mu = red[0][0] * (1.f / 1024.f);
  float var = red[1][0] * (1.f / 1024.f) - mu * mu;
  float rs = rsqrtf(var + 1e-5f);
  float o0 = (s0 - mu) * rs * g[t * 4 + 0] + be[t * 4 + 0];
  float o1 = (s1 - mu) * rs * g[t * 4 + 1] + be[t * 4 + 1];
  float o2 = (s2 - mu) * rs * g[t * 4 + 2] + be[t * 4 + 2];
  float o3 = (s3 - mu) * rs * g[t * 4 + 3] + be[t * 4 + 3];
  if (outF) {
    float4 ov = {o0, o1, o2, o3};
    *reinterpret_cast<float4*>(outF + base) = ov;
  }
  if (outB) {
    union { u16 r[4]; uint2 u; } pk;
    pk.r[0] = f2b(o0); pk.r[1] = f2b(o1); pk.r[2] = f2b(o2); pk.r[3] = f2b(o3);
    *reinterpret_cast<uint2*>(outB + base) = pk.u;
  }
}

// ---------------------------------------------------------------------------
// Orchestration
// ---------------------------------------------------------------------------
extern "C" void kernel_launch(void* const* d_in, const int* in_sizes, int n_in,
                              void* d_out, int out_size, void* d_ws, size_t ws_size,
                              hipStream_t stream) {
  const float* x      = (const float*)d_in[0];
  const float* qkv_w  = (const float*)d_in[1];
  const float* proj_w = (const float*)d_in[2];
  const float* proj_b = (const float*)d_in[3];
  const float* ln1_g  = (const float*)d_in[4];
  const float* ln1_b  = (const float*)d_in[5];
  const float* w1     = (const float*)d_in[6];
  const float* b1     = (const float*)d_in[7];
  const float* prelua = (const float*)d_in[8];
  const float* w2     = (const float*)d_in[9];
  const float* b2     = (const float*)d_in[10];
  const float* ln2_g  = (const float*)d_in[11];
  const float* ln2_b  = (const float*)d_in[12];
  (void)in_sizes; (void)n_in; (void)out_size; (void)ws_size;

  char* ws = (char*)d_ws;
  const size_t MB = 1024 * 1024;
  u16* wqkv  = (u16*)(ws + 0);         //  6 MB  qkv_w bf16
  u16* wproj = (u16*)(ws + 6 * MB);    //  2 MB
  u16* w1b   = (u16*)(ws + 8 * MB);    //  8 MB
  u16* w2b   = (u16*)(ws + 16 * MB);   //  8 MB
  u16* xb    = (u16*)(ws + 24 * MB);   // 16 MB  x bf16
  u16* qb    = (u16*)(ws + 40 * MB);   // 16 MB  [B,H,N,64]
  u16* kb    = (u16*)(ws + 56 * MB);   // 16 MB  [B,H,N,64]
  u16* vtb   = (u16*)(ws + 72 * MB);   // 16 MB  [B,H,64,N]  (dim-major)
  u16* ob    = (u16*)(ws + 88 * MB);   // 16 MB  attention out, token-major
  u16* hb    = (u16*)(ws + 40 * MB);   // 64 MB  MLP hidden, reuses qb..ob
  float* y1  = (float*)(ws + 104 * MB);// 32 MB  proj out (then MLP2 out)
  float* y2  = y1;
  float* x1f = (float*)(ws + 136 * MB);// 32 MB  LN1 out f32
  u16* x1b   = (u16*)(ws + 168 * MB);  // 16 MB  LN1 out bf16  (total 184 MB)

  auto conv = [&](const float* in, u16* out, int n) {
    int n4 = n >> 2;
    f2bf_kernel<<<(n4 + 255) / 256, 256, 0, stream>>>(in, out, n4);
  };
  conv(qkv_w, wqkv, 3072 * 1024);
  conv(proj_w, wproj, 1024 * 1024);
  conv(w1, w1b, 4096 * 1024);
  conv(w2, w2b, 1024 * 4096);
  conv(x, xb, 8192 * 1024);

  // QKV projection: [8192,1024] x [3072,1024]^T -> q,k,v (scatter epilogue)
  gemm_bf16_kernel<0><<<dim3(24, 64), 256, 0, stream>>>(
      xb, wqkv, 1024, 3072, nullptr, nullptr, nullptr, nullptr, qb, kb, vtb);

  // Flash attention -> ob [token,1024] bf16
  attn_kernel<<<dim3(64, 16), 256, 0, stream>>>(qb, kb, vtb, ob);

  // Output projection + bias -> y1 (f32)
  gemm_bf16_kernel<1><<<dim3(8, 64), 256, 0, stream>>>(
      ob, wproj, 1024, 1024, proj_b, nullptr, y1, nullptr, nullptr, nullptr, nullptr);

  // LN1(x + y1) -> x1f / x1b
  ln_kernel<<<8192, 256, 0, stream>>>(x, y1, ln1_g, ln1_b, x1f, x1b);

  // MLP up: [8192,1024] x [4096,1024]^T + b1, PReLU -> hb bf16
  gemm_bf16_kernel<2><<<dim3(32, 64), 256, 0, stream>>>(
      x1b, w1b, 1024, 4096, b1, prelua, nullptr, hb, nullptr, nullptr, nullptr);

  // MLP down: [8192,4096] x [1024,4096]^T + b2 -> y2 f32
  gemm_bf16_kernel<1><<<dim3(8, 64), 256, 0, stream>>>(
      hb, w2b, 4096, 1024, b2, nullptr, y2, nullptr, nullptr, nullptr, nullptr);

  // LN2(x1 + y2) -> d_out
  ln_kernel<<<8192, 256, 0, stream>>>(x1f, y2, ln2_g, ln2_b, (float*)d_out, nullptr);
}
